// BarrierNet_67250597920809
// MI455X (gfx1250) — compile-verified, matrix operand order checked by reading, not run on hardware
//
#include <hip/hip_runtime.h>
#include <hip/hip_bf16.h>
#include <math.h>

typedef __attribute__((ext_vector_type(16))) _Float16 v16h;
typedef __attribute__((ext_vector_type(8)))  _Float16 v8h;
typedef __attribute__((ext_vector_type(2)))  _Float16 v2h;
typedef __attribute__((ext_vector_type(8)))  float    v8f;
typedef __attribute__((ext_vector_type(4)))  unsigned int v4u;
typedef __attribute__((ext_vector_type(8)))  int          v8i;
typedef __attribute__((ext_vector_type(4)))  int          v4i;

#if __has_builtin(__builtin_amdgcn_tensor_load_to_lds) && \
    __has_builtin(__builtin_amdgcn_s_wait_tensorcnt)
#define HAVE_TDM 1
#else
#define HAVE_TDM 0
#endif

#if HAVE_TDM
__device__ __forceinline__ void tdm_issue(v4u g0, v8i g1) {
  v4i z4 = {};
#if defined(__clang_major__) && (__clang_major__ >= 23)
  v8i z8 = {};
  __builtin_amdgcn_tensor_load_to_lds(g0, g1, z4, z4, z8, 0);
#else
  __builtin_amdgcn_tensor_load_to_lds(g0, g1, z4, z4, 0);
#endif
}

// 2D tile load via Tensor Data Mover:
//   tensor: rows x width (f16, row-major, row stride = width elements)
//   tile  : tile_h rows x 32 halves, written to LDS with +16B pad per 64B line
//           (reproduces the 40-half LDS row stride used by the compute side)
__device__ __forceinline__ void tdm_load_tile(const _Float16* gtile,
                                              unsigned lds_addr,
                                              int width, int rows,
                                              int tile_h) {
  unsigned long long ga = (unsigned long long)(size_t)gtile;
  v4u g0;
  g0.x = 1u;                                   // count=1, user descriptor
  g0.y = lds_addr;                             // LDS byte address
  g0.z = (unsigned)(ga & 0xffffffffu);         // global_addr[31:0]
  g0.w = (unsigned)((ga >> 32) & 0x01ffffffu)  // global_addr[56:32]
         | 0x80000000u;                        // type=2 ("image")
  unsigned uw = (unsigned)width, ur = (unsigned)rows;
  v8i g1;
  g1[0] = (int)((1u << 16)      // data_size = 2 bytes
              | (1u << 20)      // pad_enable
              | (3u << 22)      // pad_interval: 16 DWORDs (64B line)
              | (3u << 25));    // pad_amount : 4 DWORDs (16B)
  g1[1] = (int)((uw & 0xffffu) << 16);                    // tensor_dim0[15:0]
  g1[2] = (int)((uw >> 16) | ((ur & 0xffffu) << 16));     // dim0 hi | dim1 lo
  g1[3] = (int)((ur >> 16) | (32u << 16));                // dim1 hi | tile_dim0=32
  g1[4] = (int)((unsigned)tile_h & 0xffffu);              // tile_dim1; tile_dim2=0
  g1[5] = (int)uw;                                        // tensor_dim0_stride lo
  g1[6] = 0;                                              // stride hi | dim1_stride lo
  g1[7] = 0;
  tdm_issue(g0, g1);
}
#endif  // HAVE_TDM

// ---------------------------------------------------------------------------
// Weight transpose + f32 -> f16 convert:  W (K x N) -> Wt (N x K)
// ---------------------------------------------------------------------------
__global__ void transpose_f32_to_f16(const float* __restrict__ W,
                                     _Float16* __restrict__ Wt,
                                     int K, int N) {
  int idx = blockIdx.x * blockDim.x + threadIdx.x;
  if (idx >= K * N) return;
  int n = idx / K;
  int k = idx - n * K;
  Wt[idx] = (_Float16)W[(size_t)k * N + n];
}

// ---------------------------------------------------------------------------
// Layer 1: x (M x 5, f32) @ W1 (5 x N, f32) + b1, ReLU, store f16
// ---------------------------------------------------------------------------
__global__ void layer1_kernel(const float* __restrict__ x,
                              const float* __restrict__ W1,
                              const float* __restrict__ b1,
                              _Float16* __restrict__ out,
                              int M, int N) {
  int idx = blockIdx.x * blockDim.x + threadIdx.x;
  if (idx >= M * N) return;
  int m = idx / N;
  int n = idx - m * N;
  float s = b1[n];
#pragma unroll
  for (int k = 0; k < 5; ++k)
    s += x[m * 5 + k] * W1[k * N + n];
  out[idx] = (_Float16)fmaxf(s, 0.0f);
}

// ---------------------------------------------------------------------------
// Tiled WMMA GEMM:  Out (M x N, f16) = act(A (M x K, f16) @ Wt^T + bias)
// Block tile 128x64, 8 waves (4Mx2N), wave 32x32 = 2x2 WMMA 16x16x32 tiles.
// Double-buffered LDS; tile fill by Tensor Data Mover (wave 0 issues,
// TENSORcnt + one barrier per K-step), fallback: per-thread b128 loads.
// ---------------------------------------------------------------------------
__global__ __launch_bounds__(256) void gemm_wmma_f16(
    const _Float16* __restrict__ A,
    const _Float16* __restrict__ Wt,
    const float* __restrict__ bias,
    _Float16* __restrict__ Out,
    int M, int N, int K, int relu) {
  constexpr int BM = 128, BN = 64, KT = 32, LDSS = KT + 8;  // 40-half stride
  __shared__ _Float16 As[2][BM * LDSS];
  __shared__ _Float16 Bs[2][BN * LDSS];

  const int tid  = threadIdx.x;
  const int lane = tid & 31;
  const int wave = tid >> 5;
  const int wm = (wave & 3) * 32;
  const int wn = (wave >> 2) * 32;
  const int bm = blockIdx.x * BM;
  const int bn = blockIdx.y * BN;

  v8f acc[2][2] = {};

  // fragment lane decomposition (CDNA5 16-bit A/B layout)
  const int frow = lane & 15;
  const int kgrp = (lane >> 4) << 3;

  const size_t abase = (size_t)bm * K;
  const size_t bbase = (size_t)bn * K;
  const int nsteps = K / KT;

#if HAVE_TDM
  const unsigned ldsA0 = (unsigned)(size_t)&As[0][0];
  const unsigned ldsB0 = (unsigned)(size_t)&Bs[0][0];
  if (wave == 0) {
    tdm_load_tile(A + abase, ldsA0, K, M, BM);
    tdm_load_tile(Wt + bbase, ldsB0, K, N, BN);
  }
#else
  // per-thread tile-fill mapping
  const int arow = tid >> 1;
  const int acol = (tid & 1) * 16;
  const int brow = tid >> 2;
  const int bcol = (tid & 3) * 8;
  {
    const _Float16* ag = A + abase + (size_t)arow * K + acol;
    *(v8h*)&As[0][arow * LDSS + acol]     = *(const v8h*)(ag);
    *(v8h*)&As[0][arow * LDSS + acol + 8] = *(const v8h*)(ag + 8);
    const _Float16* bg = Wt + bbase + (size_t)brow * K + bcol;
    *(v8h*)&Bs[0][brow * LDSS + bcol] = *(const v8h*)(bg);
  }
#endif

  for (int s = 0; s < nsteps; ++s) {
    const int cur = s & 1;
#if HAVE_TDM
    if (wave == 0) __builtin_amdgcn_s_wait_tensorcnt(0);
#endif
    __syncthreads();  // buffer `cur` is ready; buffer `cur^1` is free

    if (s + 1 < nsteps) {
      const int nxt = cur ^ 1;
      const int k1 = (s + 1) * KT;
#if HAVE_TDM
      if (wave == 0) {
        tdm_load_tile(A + abase + k1, ldsA0 + (unsigned)(nxt * BM * LDSS * 2),
                      K, M, BM);
        tdm_load_tile(Wt + bbase + k1, ldsB0 + (unsigned)(nxt * BN * LDSS * 2),
                      K, N, BN);
      }
#else
      const _Float16* ag = A + abase + (size_t)arow * K + k1 + acol;
      *(v8h*)&As[nxt][arow * LDSS + acol]     = *(const v8h*)(ag);
      *(v8h*)&As[nxt][arow * LDSS + acol + 8] = *(const v8h*)(ag + 8);
      const _Float16* bg = Wt + bbase + (size_t)brow * K + k1 + bcol;
      *(v8h*)&Bs[nxt][brow * LDSS + bcol] = *(const v8h*)(bg);
#endif
    }

    v16h afrag[2], bfrag[2];
#pragma unroll
    for (int t = 0; t < 2; ++t) {
      const _Float16* ap = &As[cur][(wm + t * 16 + frow) * LDSS];
      const _Float16* bp = &Bs[cur][(wn + t * 16 + frow) * LDSS];
#pragma unroll
      for (int j = 0; j < 8; ++j) {
        // VGPR j holds K pair: j<4 -> K = kgrp+2j ; j>=4 -> K = 16+kgrp+2(j-4)
        int kk = ((j >> 2) << 4) + kgrp + ((j & 3) << 1);
        v2h ap2 = *(const v2h*)(ap + kk);
        v2h bp2 = *(const v2h*)(bp + kk);
        afrag[t][2 * j]     = ap2[0];
        afrag[t][2 * j + 1] = ap2[1];
        bfrag[t][2 * j]     = bp2[0];
        bfrag[t][2 * j + 1] = bp2[1];
      }
    }

#pragma unroll
    for (int i = 0; i < 2; ++i)
#pragma unroll
      for (int j = 0; j < 2; ++j)
        acc[i][j] = __builtin_amdgcn_wmma_f32_16x16x32_f16(
            false, afrag[i], false, bfrag[j], (short)0, acc[i][j],
            false, false);
  }

  // Epilogue: C/D layout -> m = v + 8*(lane>=16), n = lane&15
  const int erow = (lane >> 4) << 3;
  const int ecol = lane & 15;
#pragma unroll
  for (int i = 0; i < 2; ++i) {
#pragma unroll
    for (int j = 0; j < 2; ++j) {
      int ng = bn + wn + j * 16 + ecol;
      float bb = bias[ng];
#pragma unroll
      for (int v = 0; v < 8; ++v) {
        int mg = bm + wm + i * 16 + erow + v;
        float val = acc[i][j][v] + bb;
        if (relu) val = fmaxf(val, 0.0f);
        Out[(size_t)mg * N + ng] = (_Float16)val;
      }
    }
  }
}

// ---------------------------------------------------------------------------
// Head: per-row 512-length dots (one wave per row) + CBF-QP scalar epilogue.
// ---------------------------------------------------------------------------
__global__ __launch_bounds__(256) void head_kernel(
    const _Float16* __restrict__ x41h, const _Float16* __restrict__ x42h,
    const float* __restrict__ W51, const float* __restrict__ b51,
    const float* __restrict__ W52, const float* __restrict__ b52,
    const float* __restrict__ x, const float* __restrict__ mean,
    const float* __restrict__ stdv, const float* __restrict__ mean_label,
    const float* __restrict__ std_label, float* __restrict__ out, int M) {
  int row  = blockIdx.x * 8 + (threadIdx.x >> 5);
  int lane = threadIdx.x & 31;
  if (row >= M) return;

  const _Float16* a = x41h + (size_t)row * 512;
  const _Float16* c = x42h + (size_t)row * 512;
  float s51a = 0.f, s51b = 0.f, s52a = 0.f, s52b = 0.f;
#pragma unroll 4
  for (int j = 0; j < 16; ++j) {
    int k = lane * 16 + j;
    float av = (float)a[k];
    float cv = (float)c[k];
    s51a += av * W51[2 * k];
    s51b += av * W51[2 * k + 1];
    s52a += cv * W52[2 * k];
    s52b += cv * W52[2 * k + 1];
  }
#pragma unroll
  for (int off = 16; off > 0; off >>= 1) {
    s51a += __shfl_xor(s51a, off, 32);
    s51b += __shfl_xor(s51b, off, 32);
    s52a += __shfl_xor(s52a, off, 32);
    s52b += __shfl_xor(s52b, off, 32);
  }
  if (lane != 0) return;

  float x51_0 = s51a + b51[0];
  float x51_1 = s51b + b51[1];
  float x52_0 = 4.0f / (1.0f + expf(-(s52a + b52[0])));
  float x52_1 = 4.0f / (1.0f + expf(-(s52b + b52[1])));

  const float L1c = 3.0f, L2c = 3.0f, OBSX = 0.0f, OBSY = 7.0f, Rc = 4.0f;
  float t1 = x[row * 5 + 0] * stdv[0] + mean[0];
  float w1 = x[row * 5 + 1] * stdv[1] + mean[1];
  float t2 = x[row * 5 + 2] * stdv[2] + mean[2];
  float w2 = x[row * 5 + 3] * stdv[3] + mean[3];
  float s1 = sinf(t1), s2 = sinf(t2), c1 = cosf(t1), c2 = cosf(t2);

  float px = L1c * c1 + L2c * c2 - OBSX;
  float py = L1c * s1 + L2c * s2 - OBSY;
  float vx = -L1c * s1 * w1 - L2c * s2 * w2;
  float vy =  L1c * c1 * w1 + L2c * c2 * w2;
  float barrier = px * px + py * py - Rc * Rc;
  float b_dot = 2.0f * (px * vx + py * vy);
  float Lf2b = 2.0f * (vx * vx + vy * vy) +
               2.0f * px * (-L1c * c1 * w1 * w1 - L2c * c2 * w2 * w2) +
               2.0f * py * (-L1c * s1 * w1 * w1 - L2c * s2 * w2 * w2);
  float G1 = -(2.0f * px * (-L1c * s1) + 2.0f * py * (L1c * c1));
  float G2 = -(2.0f * px * (-L2c * s2) + 2.0f * py * (L2c * c2));
  float h = Lf2b + (x52_0 + x52_1) * b_dot + x52_0 * x52_1 * barrier;
  float u0_0 = -x51_0, u0_1 = -x51_1;
  float viol = G1 * u0_0 + G2 * u0_1 - h;
  float lam = fmaxf(viol, 0.0f) / (G1 * G1 + G2 * G2 + 1e-12f);
  float u_0 = u0_0 - lam * G1;
  float u_1 = u0_1 - lam * G2;

  out[row * 2 + 0] = (u_0 - mean_label[0]) / std_label[0];
  out[row * 2 + 1] = (u_1 - mean_label[1]) / std_label[1];
}

// ---------------------------------------------------------------------------
extern "C" void kernel_launch(void* const* d_in, const int* in_sizes, int n_in,
                              void* d_out, int out_size, void* d_ws,
                              size_t ws_size, hipStream_t stream) {
  const float* x          = (const float*)d_in[0];
  const float* mean       = (const float*)d_in[2];
  const float* stdv       = (const float*)d_in[3];
  const float* mean_label = (const float*)d_in[4];
  const float* std_label  = (const float*)d_in[5];
  const float* W1  = (const float*)d_in[6];  const float* b1  = (const float*)d_in[7];
  const float* W2  = (const float*)d_in[8];  const float* b2  = (const float*)d_in[9];
  const float* W31 = (const float*)d_in[10]; const float* b31 = (const float*)d_in[11];
  const float* W32 = (const float*)d_in[12]; const float* b32 = (const float*)d_in[13];
  const float* W41 = (const float*)d_in[14]; const float* b41 = (const float*)d_in[15];
  const float* W42 = (const float*)d_in[16]; const float* b42 = (const float*)d_in[17];
  const float* W51 = (const float*)d_in[18]; const float* b51 = (const float*)d_in[19];
  const float* W52 = (const float*)d_in[20]; const float* b52 = (const float*)d_in[21];

  const int M = in_sizes[0] / 5;  // 8192
  float* out = (float*)d_out;

  // workspace layout (f16 buffers), 256B aligned
  char* ws = (char*)d_ws;
  size_t off = 0;
  auto alloc = [&](size_t bytes) -> void* {
    void* p = ws + off;
    off += (bytes + 255) & ~(size_t)255;
    return p;
  };
  _Float16* W2t  = (_Float16*)alloc((size_t)1024 * 1024 * 2);
  _Float16* W31t = (_Float16*)alloc((size_t)512 * 1024 * 2);
  _Float16* W32t = (_Float16*)alloc((size_t)512 * 1024 * 2);
  _Float16* W41t = (_Float16*)alloc((size_t)512 * 512 * 2);
  _Float16* W42t = (_Float16*)alloc((size_t)512 * 512 * 2);
  _Float16* x1h  = (_Float16*)alloc((size_t)M * 1024 * 2);
  _Float16* x2h  = (_Float16*)alloc((size_t)M * 1024 * 2);
  _Float16* x31h = (_Float16*)alloc((size_t)M * 512 * 2);
  _Float16* x32h = (_Float16*)alloc((size_t)M * 512 * 2);
  _Float16* x41h = (_Float16*)alloc((size_t)M * 512 * 2);
  _Float16* x42h = (_Float16*)alloc((size_t)M * 512 * 2);

  // 1. weight transposes (f32 -> f16 N x K)
  transpose_f32_to_f16<<<(1024 * 1024 + 255) / 256, 256, 0, stream>>>(W2, W2t, 1024, 1024);
  transpose_f32_to_f16<<<(1024 * 512 + 255) / 256, 256, 0, stream>>>(W31, W31t, 1024, 512);
  transpose_f32_to_f16<<<(1024 * 512 + 255) / 256, 256, 0, stream>>>(W32, W32t, 1024, 512);
  transpose_f32_to_f16<<<(512 * 512 + 255) / 256, 256, 0, stream>>>(W41, W41t, 512, 512);
  transpose_f32_to_f16<<<(512 * 512 + 255) / 256, 256, 0, stream>>>(W42, W42t, 512, 512);

  // 2. layer 1 (K=5, VALU)
  layer1_kernel<<<(M * 1024 + 255) / 256, 256, 0, stream>>>(x, W1, b1, x1h, M, 1024);

  // 3. WMMA GEMM chain
  dim3 blk(256);
  gemm_wmma_f16<<<dim3(M / 128, 1024 / 64), blk, 0, stream>>>(x1h,  W2t,  b2,  x2h,  M, 1024, 1024, 1);
  gemm_wmma_f16<<<dim3(M / 128,  512 / 64), blk, 0, stream>>>(x2h,  W31t, b31, x31h, M,  512, 1024, 1);
  gemm_wmma_f16<<<dim3(M / 128,  512 / 64), blk, 0, stream>>>(x2h,  W32t, b32, x32h, M,  512, 1024, 1);
  gemm_wmma_f16<<<dim3(M / 128,  512 / 64), blk, 0, stream>>>(x31h, W41t, b41, x41h, M,  512,  512, 1);
  gemm_wmma_f16<<<dim3(M / 128,  512 / 64), blk, 0, stream>>>(x32h, W42t, b42, x42h, M,  512,  512, 1);

  // 4. head + CBF-QP epilogue
  head_kernel<<<(M + 7) / 8, 256, 0, stream>>>(x41h, x42h, W51, b51, W52, b52,
                                               x, mean, stdv, mean_label,
                                               std_label, out, M);
}
